// GAT_16707422781830
// MI455X (gfx1250) — matrix operand order repaired and checked by reading, not compiled, not run on hardware
//
#include <hip/hip_runtime.h>
#include <hip/hip_bf16.h>
#include <math.h>

typedef __attribute__((ext_vector_type(16))) __bf16 v16bf;
typedef __attribute__((ext_vector_type(8)))  float  v8f;

#define NEG_SLOPE 0.01f

// ---------- helpers ----------
__device__ __forceinline__ unsigned short f2bf(float f) {
    unsigned u = __float_as_uint(f);
    unsigned r = (u + 0x7FFFu + ((u >> 16) & 1u)) >> 16;   // RNE
    return (unsigned short)r;
}

// ---------- fill / convert ----------
__global__ void k_fill_f32(float* __restrict__ p, float v, long long n) {
    long long i = (long long)blockIdx.x * blockDim.x + threadIdx.x;
    long long s = (long long)gridDim.x * blockDim.x;
    for (; i < n; i += s) p[i] = v;
}
__global__ void k_fill_u32(unsigned* __restrict__ p, unsigned v, long long n) {
    long long i = (long long)blockIdx.x * blockDim.x + threadIdx.x;
    long long s = (long long)gridDim.x * blockDim.x;
    for (; i < n; i += s) p[i] = v;
}
__global__ void k_f32_to_bf16(const float* __restrict__ in, unsigned short* __restrict__ out, long long n) {
    long long i = (long long)blockIdx.x * blockDim.x + threadIdx.x;
    long long s = (long long)gridDim.x * blockDim.x;
    for (; i < n; i += s) out[i] = f2bf(in[i]);
}
__global__ void k_tanh_to_bf16(const float* __restrict__ in, unsigned short* __restrict__ out, long long n) {
    long long i = (long long)blockIdx.x * blockDim.x + threadIdx.x;
    long long s = (long long)gridDim.x * blockDim.x;
    for (; i < n; i += s) out[i] = f2bf(tanhf(in[i]));
}
__global__ void k_elu_to_bf16(const float* __restrict__ in, unsigned short* __restrict__ out, long long n) {
    long long i = (long long)blockIdx.x * blockDim.x + threadIdx.x;
    long long s = (long long)gridDim.x * blockDim.x;
    for (; i < n; i += s) {
        float x = in[i];
        out[i] = f2bf(x >= 0.f ? x : (__expf(x) - 1.f));
    }
}

// ---------- BF16 WMMA GEMM: C(MxNout) = A(MxK) * B(NoutxK)^T ----------
// A: bf16 row-major (lda=K), B: bf16 row-major weights W[nout][k] (ldb=K), C: f32.
// One wave -> 16x64 output strip (4 n-tiles). Software-pipelined K loop:
// next iteration's fragments are loaded before current WMMAs so the wait
// before the WMMA group only covers loads from one iteration earlier.
// Requires M%16==0, Nout%64==0, K%32==0 (K>=64).
__global__ void k_wmma_gemm(const unsigned short* __restrict__ A,
                            const unsigned short* __restrict__ B,
                            float* __restrict__ C,
                            int M, int K, int Nout, int jobs, int ngroups)
{
    int w    = threadIdx.x >> 5;
    int lane = threadIdx.x & 31;
    int gw   = blockIdx.x * (blockDim.x >> 5) + w;
    if (gw >= jobs) return;                 // wave-uniform: EXEC all-ones inside
    int mtile = gw / ngroups;
    int ngrp  = gw % ngroups;
    int lo = lane & 15;
    int hi = lane >> 4;

    const unsigned short* arow = A + (size_t)(mtile * 16 + lo) * K + hi * 8;
    const unsigned short* brow[4];
#pragma unroll
    for (int j = 0; j < 4; ++j)
        brow[j] = B + (size_t)((ngrp * 4 + j) * 16 + lo) * K + hi * 8;

    v8f acc[4];
#pragma unroll
    for (int j = 0; j < 4; ++j)
#pragma unroll
        for (int r = 0; r < 8; ++r) acc[j][r] = 0.f;

    union F { v16bf v; uint4 u[2]; };
    F af, bf[4];

    // prologue: k0 = 0
    af.u[0] = *(const uint4*)(arow);
    af.u[1] = *(const uint4*)(arow + 16);
#pragma unroll
    for (int j = 0; j < 4; ++j) {
        bf[j].u[0] = *(const uint4*)(brow[j]);
        bf[j].u[1] = *(const uint4*)(brow[j] + 16);
    }

    int kiters = K >> 5;
    for (int it = 0; it < kiters - 1; ++it) {
        int kn = (it + 1) << 5;
        F an, bn[4];
        // prefetch next iteration's fragments
        an.u[0] = *(const uint4*)(arow + kn);
        an.u[1] = *(const uint4*)(arow + kn + 16);
#pragma unroll
        for (int j = 0; j < 4; ++j) {
            bn[j].u[0] = *(const uint4*)(brow[j] + kn);
            bn[j].u[1] = *(const uint4*)(brow[j] + kn + 16);
        }
        // consume current fragments
#pragma unroll
        for (int j = 0; j < 4; ++j)
            acc[j] = __builtin_amdgcn_wmma_f32_16x16x32_bf16(
                false, af.v, false, bf[j].v, (short)0, acc[j], false, false);
        af = an;
#pragma unroll
        for (int j = 0; j < 4; ++j) bf[j] = bn[j];
    }
    // epilogue
#pragma unroll
    for (int j = 0; j < 4; ++j)
        acc[j] = __builtin_amdgcn_wmma_f32_16x16x32_bf16(
            false, af.v, false, bf[j].v, (short)0, acc[j], false, false);

#pragma unroll
    for (int j = 0; j < 4; ++j) {
        int ncol = (ngrp * 4 + j) * 16 + lo;
#pragma unroll
        for (int r = 0; r < 8; ++r) {
            int row = mtile * 16 + r + 8 * hi;
            C[(size_t)row * Nout + ncol] = acc[j][r];
        }
    }
}

// ---------- per-node attention scalars: s_src = z . a[:D], s_dst = z . a[D:] ----------
__global__ void k_scores(const float* __restrict__ Z, const float* __restrict__ Avec,
                         float* __restrict__ s_src, float* __restrict__ s_dst,
                         int Nnodes, int D)
{
    int w    = threadIdx.x >> 5;
    int lane = threadIdx.x & 31;
    int node = blockIdx.x * (blockDim.x >> 5) + w;
    if (node >= Nnodes) return;
    const float* z = Z + (size_t)node * D;
    float ss = 0.f, sd = 0.f;
    for (int d = lane; d < D; d += 32) {
        float zv = z[d];
        ss += zv * Avec[d];
        sd += zv * Avec[D + d];
    }
#pragma unroll
    for (int off = 16; off; off >>= 1) {
        ss += __shfl_xor(ss, off, 32);
        sd += __shfl_xor(sd, off, 32);
    }
    if (lane == 0) { s_src[node] = ss; s_dst[node] = sd; }
}

// ---------- edge: e = leaky(s_src[src]+s_dst[dst]); segment max via monotonic-uint atomicMax ----------
__global__ void k_edge_max(const float* __restrict__ s_src, const float* __restrict__ s_dst,
                           const int* __restrict__ src, const int* __restrict__ dst,
                           float* __restrict__ e_out, unsigned* __restrict__ emax_bits, int E)
{
    int i = blockIdx.x * blockDim.x + threadIdx.x;
    if (i >= E) return;
    float e = s_src[src[i]] + s_dst[dst[i]];
    e = (e >= 0.f) ? e : NEG_SLOPE * e;
    e_out[i] = e;
    unsigned u = __float_as_uint(e);
    u = (u & 0x80000000u) ? ~u : (u | 0x80000000u);   // order-preserving encoding
    atomicMax(&emax_bits[dst[i]], u);
}

// decode; bits==0 (no incoming edge) decodes to NaN -> 0, matching the reference
__global__ void k_emax_fin(const unsigned* __restrict__ bits, float* __restrict__ emax, int N)
{
    int i = blockIdx.x * blockDim.x + threadIdx.x;
    if (i >= N) return;
    unsigned u = bits[i];
    unsigned b = (u & 0x80000000u) ? (u ^ 0x80000000u) : ~u;
    float f = __uint_as_float(b);
    if (!isfinite(f)) f = 0.f;
    emax[i] = f;
}

__global__ void k_edge_exp(const float* __restrict__ e, const int* __restrict__ dst,
                           const float* __restrict__ emax,
                           float* __restrict__ ex, float* __restrict__ denom, int E)
{
    int i = blockIdx.x * blockDim.x + threadIdx.x;
    if (i >= E) return;
    int d = dst[i];
    float x = __expf(e[i] - emax[d]);
    ex[i] = x;
    atomicAdd(&denom[d], x);
}

// ---------- out[dst] += alpha * z[src] : one wave per edge ----------
__global__ void k_aggregate(const float* __restrict__ Z, const float* __restrict__ ex,
                            const float* __restrict__ denom,
                            const int* __restrict__ src, const int* __restrict__ dst,
                            float* __restrict__ out, int E, int D)
{
    int w    = threadIdx.x >> 5;
    int lane = threadIdx.x & 31;
    int e = blockIdx.x * (blockDim.x >> 5) + w;
    if (e >= E) return;
    int s = src[e], d = dst[e];
    float alpha = ex[e] / fmaxf(denom[d], 1e-9f);
    const float* zs = Z + (size_t)s * D;
    float*       od = out + (size_t)d * D;
    for (int c = lane; c < D; c += 32)
        atomicAdd(&od[c], alpha * zs[c]);
}

// ---------------------------------------------------------------------------
static inline int cdiv(long long a, long long b) { return (int)((a + b - 1) / b); }

extern "C" void kernel_launch(void* const* d_in, const int* in_sizes, int n_in,
                              void* d_out, int out_size, void* d_ws, size_t ws_size,
                              hipStream_t stream)
{
    const float* h   = (const float*)d_in[0];
    const int*   src = (const int*)d_in[1];
    const int*   dst = (const int*)d_in[2];
    const float* W1  = (const float*)d_in[4];
    const float* A1  = (const float*)d_in[5];
    const float* W2  = (const float*)d_in[6];
    const float* A2  = (const float*)d_in[7];
    const float* W3  = (const float*)d_in[8];
    const float* A3  = (const float*)d_in[9];

    const int HID = in_sizes[5] / 2;          // A1 = (1, 2*HID)
    const int IN  = in_sizes[4] / HID;        // W1 = (HID, IN)
    const int N   = in_sizes[0] / IN;         // h  = (N, IN)
    const int OUT = in_sizes[9] / 2;          // A3 = (1, 2*OUT)
    const int E   = in_sizes[1];

    // ---- workspace carve ----
    char* base = (char*)d_ws; size_t off = 0;
    auto carve = [&](size_t bytes) -> char* {
        char* p = base + off; off = (off + bytes + 255) & ~(size_t)255; return p;
    };
    int DMAX = IN > HID ? IN : HID;
    unsigned short* xbf  = (unsigned short*)carve((size_t)N * DMAX * 2);
    unsigned short* w1bf = (unsigned short*)carve((size_t)HID * IN * 2);
    unsigned short* w2bf = (unsigned short*)carve((size_t)HID * HID * 2);
    unsigned short* w3bf = (unsigned short*)carve((size_t)OUT * HID * 2);
    float*    z     = (float*)carve((size_t)N * HID * 4);
    float*    agg   = (float*)carve((size_t)N * HID * 4);
    float*    ssrc  = (float*)carve((size_t)N * 4);
    float*    sdst  = (float*)carve((size_t)N * 4);
    unsigned* emaxb = (unsigned*)carve((size_t)N * 4);
    float*    emaxf = (float*)carve((size_t)N * 4);
    float*    den   = (float*)carve((size_t)N * 4);
    float*    ebuf  = (float*)carve((size_t)E * 4);
    float*    exbuf = (float*)carve((size_t)E * 4);

    const int T = 256;                       // 8 waves / block
    const int CG = 2048;                     // grid-stride grids for elementwise

    // ---- bf16 conversions (once per call) ----
    k_f32_to_bf16<<<CG, T, 0, stream>>>(h,  xbf,  (long long)N * IN);
    k_f32_to_bf16<<<CG, T, 0, stream>>>(W1, w1bf, (long long)HID * IN);
    k_f32_to_bf16<<<CG, T, 0, stream>>>(W2, w2bf, (long long)HID * HID);
    k_f32_to_bf16<<<CG, T, 0, stream>>>(W3, w3bf, (long long)OUT * HID);

    auto run_layer = [&](const unsigned short* Xbf, const unsigned short* Wbf,
                         const float* Avec, int Din, int Dout, float* aggdst) {
        // GEMM: z = Xbf * Wbf^T   (N x Dout)
        int mtiles  = N / 16;                // N = 50000 = 3125*16
        int ngroups = Dout / 64;             // 64-wide strips per wave
        int jobs    = mtiles * ngroups;
        k_wmma_gemm<<<cdiv(jobs, 8), T, 0, stream>>>(Xbf, Wbf, z, N, Din, Dout, jobs, ngroups);
        // per-node score scalars
        k_scores<<<cdiv(N, 8), T, 0, stream>>>(z, Avec, ssrc, sdst, N, Dout);
        // segment softmax over dst
        k_fill_u32<<<CG, T, 0, stream>>>(emaxb, 0u, N);
        k_fill_f32<<<CG, T, 0, stream>>>(den, 0.f, N);
        k_edge_max<<<cdiv(E, T), T, 0, stream>>>(ssrc, sdst, src, dst, ebuf, emaxb, E);
        k_emax_fin<<<cdiv(N, T), T, 0, stream>>>(emaxb, emaxf, N);
        k_edge_exp<<<cdiv(E, T), T, 0, stream>>>(ebuf, dst, emaxf, exbuf, den, E);
        // weighted aggregation
        k_fill_f32<<<CG, T, 0, stream>>>(aggdst, 0.f, (long long)N * Dout);
        k_aggregate<<<cdiv(E, 8), T, 0, stream>>>(z, exbuf, den, src, dst, aggdst, E, Dout);
    };

    // Layer 1: IN -> HID, tanh
    run_layer(xbf, w1bf, A1, IN, HID, agg);
    k_tanh_to_bf16<<<CG, T, 0, stream>>>(agg, xbf, (long long)N * HID);

    // Layer 2: HID -> HID, elu
    run_layer(xbf, w2bf, A2, HID, HID, agg);
    k_elu_to_bf16<<<CG, T, 0, stream>>>(agg, xbf, (long long)N * HID);

    // Layer 3: HID -> OUT, straight into d_out (f32)
    run_layer(xbf, w3bf, A3, HID, OUT, (float*)d_out);
}
